// Self_Attention_30193620091602
// MI455X (gfx1250) — matrix operand order repaired and checked by reading, not compiled, not run on hardware
//
#include <hip/hip_runtime.h>

// ---------------------------------------------------------------------------
// MI455X (gfx1250) fused self-attention pipeline, all GEMMs via
// v_wmma_f32_16x16x32_f16 (wave32). fp32 inputs are converted to f16 in
// registers / once for weights; accumulation is fp32.
// Q and V projections are fused (X read once); f16 A-tiles are staged to LDS
// with GLOBAL_LOAD_ASYNC_TO_LDS_B128 (ASYNCcnt).
// ---------------------------------------------------------------------------

typedef __attribute__((ext_vector_type(16))) _Float16 v16h;
typedef __attribute__((ext_vector_type(8)))  _Float16 v8h;
typedef __attribute__((ext_vector_type(8)))  float    v8f;
typedef __attribute__((ext_vector_type(4)))  float    v4f;

union V16U { v16h v; v8h h[2]; };

#define LDS_STRIDE 40   // 128x32 f16 tile, padded: 80B rows -> conflict-free b128 reads

// ---------------------------------------------------------------------------
// Kernel 0: convert Wq, Wv, Wo (each 256*1024 = 262144 fp32) to f16
// ---------------------------------------------------------------------------
__global__ __launch_bounds__(256) void k_convert_w(
    const float* __restrict__ Wq, const float* __restrict__ Wv,
    const float* __restrict__ Wo,
    _Float16* __restrict__ WqH, _Float16* __restrict__ WvH,
    _Float16* __restrict__ WoH) {
  int i = blockIdx.x * 256 + threadIdx.x;
  WqH[i] = (_Float16)Wq[i];
  WvH[i] = (_Float16)Wv[i];
  WoH[i] = (_Float16)Wo[i];
}

// ---------------------------------------------------------------------------
// Kernel 1: klinT[n][b] = sum_a x[b,n,a] * Wk[a]   (one wave per output)
// ---------------------------------------------------------------------------
__global__ __launch_bounds__(256) void k_klin(
    const float* __restrict__ X, const float* __restrict__ Wk,
    _Float16* __restrict__ klinT) {
  int wid  = (blockIdx.x * 256 + threadIdx.x) >> 5;  // flat row of X: b*256+n
  int lane = threadIdx.x & 31;
  const float* row = X + (size_t)wid * 1024;
  float s = 0.f;
#pragma unroll 4
  for (int a = lane; a < 1024; a += 32) s += row[a] * Wk[a];
#pragma unroll
  for (int off = 16; off >= 1; off >>= 1) s += __shfl_xor(s, off, 32);
  if (lane == 0) {
    int b = wid >> 8, n = wid & 255;
    klinT[n * 256 + b] = (_Float16)s;
  }
}

// ---------------------------------------------------------------------------
// Kernel 2 (fused QV): Q = X*Wq^T + bq -> Qt[b][q][m]   (batch-transposed)
//                      V = X*Wv^T + bv -> V[b][n][m]    (natural)
//   X (fp32, 65536 x 1024) is staged fp32->f16 in LDS ONCE and multiplied
//   against both weight matrices: halves the dominant HBM stream.
//   256 threads = 8 waves; tile 128(M) x 64(N); 8 fp32 accumulators/wave.
// ---------------------------------------------------------------------------
__global__ __launch_bounds__(256) void k_gemm_qv(
    const float* __restrict__ X,
    const _Float16* __restrict__ WqH, const _Float16* __restrict__ WvH,
    const float* __restrict__ bq, const float* __restrict__ bv,
    _Float16* __restrict__ Qt, _Float16* __restrict__ Vout, int K) {
  __shared__ _Float16 tileA[128 * LDS_STRIDE];

  const int tid  = threadIdx.x;
  const int lane = tid & 31;
  const int wave = tid >> 5;
  const int half = lane >> 4;
  const int lcol = lane & 15;
  const int nIdx = wave & 3;   // 4 N-subtiles of 16
  const int mIdx = wave >> 2;  // 2 M-halves of 64
  const int m0 = blockIdx.x * 128;
  const int n0 = blockIdx.y * 64;

  v8f accQ[4] = {};
  v8f accV[4] = {};

  const int rl  = tid >> 1;          // staging: row 0..127
  const int kk0 = (tid & 1) * 16;    // staging: k-offset 0 or 16

  for (int k = 0; k < K; k += 32) {
    // ---- stage X tile (128 x 32) into LDS as f16 ----
    {
      const v4f* src = (const v4f*)(X + (size_t)(m0 + rl) * K + k + kk0);
      if (k + 32 < K)
        __builtin_prefetch(X + (size_t)(m0 + rl) * K + k + 32 + kk0, 0, 3);
      v8h h0, h1;
#pragma unroll
      for (int c = 0; c < 2; ++c) {
        v4f f0 = src[2 * c], f1 = src[2 * c + 1];
        v8h h;
#pragma unroll
        for (int j = 0; j < 4; ++j) { h[j] = (_Float16)f0[j]; h[4 + j] = (_Float16)f1[j]; }
        if (c == 0) h0 = h; else h1 = h;
      }
      v8h* dst = (v8h*)&tileA[rl * LDS_STRIDE + kk0];
      dst[0] = h0; dst[1] = h1;
    }
    __syncthreads();

    // ---- B fragments (N x K weights, lane reads 16 contiguous f16) ----
    const size_t boff = (size_t)(n0 + nIdx * 16 + lcol) * K + k + 16 * half;
    const v16h bq16 = *(const v16h*)(WqH + boff);
    const v16h bv16 = *(const v16h*)(WvH + boff);

    // ---- preload all A fragments, then issue 8 WMMAs back-to-back ----
    V16U a[4];
#pragma unroll
    for (int t = 0; t < 4; ++t) {
      const int ar = mIdx * 64 + t * 16 + lcol;
      a[t].h[0] = *(const v8h*)&tileA[ar * LDS_STRIDE + 16 * half];
      a[t].h[1] = *(const v8h*)&tileA[ar * LDS_STRIDE + 16 * half + 8];
    }
#pragma unroll
    for (int t = 0; t < 4; ++t)
      accQ[t] = __builtin_amdgcn_wmma_f32_16x16x32_f16(
          false, a[t].v, false, bq16, (short)0, accQ[t], false, false);
#pragma unroll
    for (int t = 0; t < 4; ++t)
      accV[t] = __builtin_amdgcn_wmma_f32_16x16x32_f16(
          false, a[t].v, false, bv16, (short)0, accV[t], false, false);
    __syncthreads();
  }

  // ---- epilogues ----
  const int col = n0 + nIdx * 16 + lcol;
  const float bqv = bq[col];
  const float bvv = bv[col];
#pragma unroll
  for (int t = 0; t < 4; ++t) {
#pragma unroll
    for (int r = 0; r < 8; ++r) {
      const int row = m0 + mIdx * 64 + t * 16 + r + 8 * half;
      const int b = row >> 8, m = row & 255;
      // Qt[b][col][m]  (B-operand layout for the att GEMM)
      Qt[((size_t)(b * 256 + col) << 8) + m] = (_Float16)(accQ[t][r] + bqv);
      // V[b][m][col]   (A-operand layout for the out GEMM)
      Vout[((size_t)row << 8) + col] = (_Float16)(accV[t][r] + bvv);
    }
  }
}

// ---------------------------------------------------------------------------
// Kernel 3: generic batched f16 GEMM: D = A(M x K) * B(N x K)^T (+bias)
//   A row-major K-contiguous, B is N x K (pre-transposed), per-batch strides.
//   A tile staged to LDS via GLOBAL_LOAD_ASYNC_TO_LDS_B128 (no VGPR round
//   trip), completion via s_wait_asynccnt. outF32: write fp32 (final output).
//   NOTE: tileA is the only __shared__ object -> LDS offset 0.
// ---------------------------------------------------------------------------
__global__ __launch_bounds__(256) void k_gemm_hh(
    const _Float16* __restrict__ A, long long aStr,
    const _Float16* __restrict__ B, long long bStr,
    const float* __restrict__ bias, void* __restrict__ out, long long oStr,
    int ldo, int outF32, int K) {
  __shared__ _Float16 tileA[128 * LDS_STRIDE];

  const int tid  = threadIdx.x;
  const int lane = tid & 31;
  const int wave = tid >> 5;
  const int half = lane >> 4;
  const int lcol = lane & 15;
  const int nIdx = wave & 3;
  const int mIdx = wave >> 2;
  const int m0 = blockIdx.x * 128;
  const int n0 = blockIdx.y * 64;
  const int batch = blockIdx.z;

  const _Float16* Ab = A + (size_t)batch * aStr;
  const _Float16* Bb = B + (size_t)batch * bStr;

  v8f acc[4] = {};

  const int rl  = tid >> 1;
  const int kk0 = (tid & 1) * 16;
  const unsigned ldsoff =
      (unsigned)((rl * LDS_STRIDE + kk0) * sizeof(_Float16));  // tileA @ LDS 0

  for (int k = 0; k < K; k += 32) {
    // ---- async DMA: 32 B per lane, global -> LDS, bypassing VGPRs ----
    {
      const _Float16* gsrc = Ab + (size_t)(m0 + rl) * K + k + kk0;
      if (k + 32 < K)
        __builtin_prefetch(Ab + (size_t)(m0 + rl) * K + k + 32 + kk0, 0, 3);
      asm volatile(
          "global_load_async_to_lds_b128 %0, %1, off\n\t"
          "global_load_async_to_lds_b128 %0, %1, off offset:16\n\t"
          "s_wait_asynccnt 0x0"
          :: "v"(ldsoff), "v"(gsrc) : "memory");
    }
    __syncthreads();

    const v16h bfrag =
        *(const v16h*)(Bb + (size_t)(n0 + nIdx * 16 + lcol) * K + k + 16 * half);

    V16U a[4];
#pragma unroll
    for (int t = 0; t < 4; ++t) {
      const int ar = mIdx * 64 + t * 16 + lcol;
      a[t].h[0] = *(const v8h*)&tileA[ar * LDS_STRIDE + 16 * half];
      a[t].h[1] = *(const v8h*)&tileA[ar * LDS_STRIDE + 16 * half + 8];
    }
#pragma unroll
    for (int t = 0; t < 4; ++t)
      acc[t] = __builtin_amdgcn_wmma_f32_16x16x32_f16(
          false, a[t].v, false, bfrag, (short)0, acc[t], false, false);
    __syncthreads();
  }

  const int col = n0 + nIdx * 16 + lcol;
  const float bval = bias ? bias[col] : 0.f;
#pragma unroll
  for (int t = 0; t < 4; ++t) {
#pragma unroll
    for (int r = 0; r < 8; ++r) {
      const int row = m0 + mIdx * 64 + t * 16 + r + 8 * half;
      const float v = acc[t][r] + bval;
      const size_t addr = (size_t)batch * oStr + (size_t)row * ldo + col;
      if (outF32) ((float*)out)[addr] = v;
      else        ((_Float16*)out)[addr] = (_Float16)v;
    }
  }
}

// ---------------------------------------------------------------------------
// Kernel 4: softmax over last axis of attP (b,n,256), write transposed
//   attT[b][q][n] for the next GEMM's B operand. One wave per row.
// ---------------------------------------------------------------------------
__global__ __launch_bounds__(256) void k_softmax(
    const _Float16* __restrict__ attP, _Float16* __restrict__ attT) {
  const int wid  = (blockIdx.x * 256 + threadIdx.x) >> 5;  // row = b*256+n
  const int lane = threadIdx.x & 31;
  const v8h h = *(const v8h*)(attP + (size_t)wid * 256 + lane * 8);
  float vals[8];
  float mx = -3.4e38f;
#pragma unroll
  for (int j = 0; j < 8; ++j) { vals[j] = (float)h[j]; mx = fmaxf(mx, vals[j]); }
#pragma unroll
  for (int off = 16; off >= 1; off >>= 1) mx = fmaxf(mx, __shfl_xor(mx, off, 32));
  float s = 0.f;
#pragma unroll
  for (int j = 0; j < 8; ++j) { vals[j] = __expf(vals[j] - mx); s += vals[j]; }
#pragma unroll
  for (int off = 16; off >= 1; off >>= 1) s += __shfl_xor(s, off, 32);
  const float inv = 1.f / s;
  const int b = wid >> 8, n = wid & 255;
#pragma unroll
  for (int j = 0; j < 8; ++j) {
    const int q = lane * 8 + j;
    attT[((size_t)(b * 256 + q) << 8) + n] = (_Float16)(vals[j] * inv);
  }
}

// ---------------------------------------------------------------------------
// Host-side launch
// ---------------------------------------------------------------------------
extern "C" void kernel_launch(void* const* d_in, const int* in_sizes, int n_in,
                              void* d_out, int out_size, void* d_ws, size_t ws_size,
                              hipStream_t stream) {
  (void)in_sizes; (void)n_in; (void)out_size; (void)ws_size;
  const float* x  = (const float*)d_in[0];  // (256,256,1024)
  const float* Wq = (const float*)d_in[1];  // (256,1024)
  const float* bq = (const float*)d_in[2];
  const float* Wk = (const float*)d_in[3];  // (1,1024)
  const float* Wv = (const float*)d_in[5];
  const float* bv = (const float*)d_in[6];
  const float* Wo = (const float*)d_in[7];  // (1024,256)
  const float* bo = (const float*)d_in[8];

  char* ws = (char*)d_ws;
  const size_t SZ_W   = 256ull * 1024 * sizeof(_Float16);     // 512 KB
  const size_t SZ_KT  = 256ull * 256 * sizeof(_Float16);      // 128 KB
  const size_t SZ_BIG = 65536ull * 256 * sizeof(_Float16);    // 32 MB

  _Float16* WqH   = (_Float16*)(ws);
  _Float16* WvH   = (_Float16*)(ws + SZ_W);
  _Float16* WoH   = (_Float16*)(ws + 2 * SZ_W);
  _Float16* klinT = (_Float16*)(ws + 3 * SZ_W);
  size_t off = ((3 * SZ_W + SZ_KT + 255) & ~(size_t)255);
  _Float16* Qt    = (_Float16*)(ws + off); off += SZ_BIG;  // Qt[b][q][m]
  _Float16* Vbf   = (_Float16*)(ws + off); off += SZ_BIG;  // V[b][n][m]
  _Float16* attP  = (_Float16*)(ws + off); off += SZ_BIG;  // pre-softmax (b,n,q)
  _Float16* attT  = (_Float16*)(ws + off); off += SZ_BIG;  // attT[b][q][n]
  _Float16* outH  = (_Float16*)(ws + off); off += SZ_BIG;  // out (b,n,q)

  // 0) weights -> f16
  k_convert_w<<<1024, 256, 0, stream>>>(Wq, Wv, Wo, WqH, WvH, WoH);
  // 1) klinT[n][b] = x . Wk
  k_klin<<<8192, 256, 0, stream>>>(x, Wk, klinT);
  // 2) fused: Q -> Qt[b][q][m], V -> V[b][n][m]   (X streamed once)
  k_gemm_qv<<<dim3(512, 4), 256, 0, stream>>>(x, WqH, WvH, bq, bv, Qt, Vbf, 1024);
  // 3) att_pre[b][n][q] = sum_m klinT[n][m] * Qt[b][q][m]
  k_gemm_hh<<<dim3(2, 4, 256), 256, 0, stream>>>(
      klinT, 0, Qt, 65536, nullptr, attP, 65536, 256, 0, 256);
  // 4) softmax rows -> attT[b][q][n]
  k_softmax<<<8192, 256, 0, stream>>>(attP, attT);
  // 5) out[b][n][q] = sum_m V[b][n][m] * attT[b][q][m]
  k_gemm_hh<<<dim3(2, 4, 256), 256, 0, stream>>>(
      Vbf, 65536, attT, 65536, nullptr, outH, 65536, 256, 0, 256);
  // 6) y[b,n,a] = sum_v out[b][n][v] * Wo[a][v] + bo  -> fp32 d_out
  k_gemm_hh<<<dim3(512, 16, 1), 256, 0, stream>>>(
      outH, 0, WoH, 0, bo, d_out, 0, 1024, 1, 256);
}
